// AttentionBlock_12077448036680
// MI455X (gfx1250) — matrix operand order repaired and checked by reading, not compiled, not run on hardware
//
#include <hip/hip_runtime.h>

typedef __attribute__((ext_vector_type(16))) __bf16 v16bf;
typedef __attribute__((ext_vector_type(8)))  float  v8f;

#define HD   256   // hidden channels C
#define AD   256   // ATTN_DIM
#define NB   16
#define NS   32
#define HW   256   // H*W
#define KTOT 512   // 2*C
#define LDK  40    // padded k-stride (bf16 elems): 80B, 16B-aligned

__device__ __forceinline__ unsigned short f32_to_bf16(float x) {
    unsigned u = __float_as_uint(x);
    unsigned r = u + 0x7FFFu + ((u >> 16) & 1u);   // round-to-nearest-even
    return (unsigned short)(r >> 16);
}
__device__ __forceinline__ float bf16_to_f32(unsigned short h) {
    return __uint_as_float(((unsigned)h) << 16);
}
__device__ __forceinline__ float tanh_fast(float x) {
#if __has_builtin(__builtin_amdgcn_tanhf)
    return __builtin_amdgcn_tanhf(x);
#elif __has_builtin(__builtin_amdgcn_tanh_f32)
    return __builtin_amdgcn_tanh_f32(x);
#else
    float e = __expf(2.0f * x);
    return 1.0f - 2.0f / (e + 1.0f);
#endif
}
// pack two f32 into one dword of two bf16 (low = first, matching ascending-k layout)
__device__ __forceinline__ unsigned pack_bf16x2(float x0, float x1) {
    return (unsigned)f32_to_bf16(x0) | ((unsigned)f32_to_bf16(x1) << 16);
}

// ---------------------------------------------------------------------------
// Kernel 0: pre-swizzle W1 (256x512 f32) into WMMA A-fragment layout, split
// into bf16 hi + lo planes. One block per (atile, kstep); 32 lanes.
// A-matrix 16-bit layout: lane m = lane&15; element j -> VGPR j>>1, half j&1;
// here k = 16*(lane>>4) + j (contiguous order; the B fragments use the same).
// ---------------------------------------------------------------------------
__global__ __launch_bounds__(32) void prep_w1_frags(
    const float* __restrict__ W1,
    unsigned short* __restrict__ fragHi,
    unsigned short* __restrict__ fragLo)
{
    int blk = blockIdx.x;                 // atile*16 + ks
    int atile = blk >> 4, ks = blk & 15;
    int lane = threadIdx.x;
    int m = lane & 15, g = lane >> 4;
    int base = (blk * 32 + lane) * 16;
#pragma unroll
    for (int j = 0; j < 16; ++j) {
        int v = j >> 1, h = j & 1;
        int k = 16 * g + j;
        int a = atile * 16 + m;
        int c = ks * 32 + k;
        float x = W1[a * KTOT + c];
        unsigned short hi = f32_to_bf16(x);
        unsigned short lo = f32_to_bf16(x - bf16_to_f32(hi));
        fragHi[base + j] = hi;
        fragLo[base + j] = lo;
    }
}

// ---------------------------------------------------------------------------
// Load one chunk's thread-portion: rows (2*kpair, 2*kpair+1), 8 pixels.
// Chunk boundary is uniform: ks<8 -> hidden, ks>=8 -> enc.
// ---------------------------------------------------------------------------
__device__ __forceinline__ void load_chunk(
    const float* __restrict__ hidden, const float* __restrict__ enc,
    int b, int s, int ph, int ks, int kpair, int cseg, float* __restrict__ f)
{
    const float* base = (ks < 8)
        ? (hidden + ((size_t)(b * HD) + ks * 32) * HW)
        : (enc + ((size_t)((b * NS + s) * HD) + (ks - 8) * 32) * HW);
    const float* src = base + (size_t)(2 * kpair) * HW + ph * 128 + cseg * 8;
    const float4* s0 = (const float4*)src;
    const float4* s1 = (const float4*)(src + HW);
    float4 a0 = s0[0], a1 = s0[1];
    float4 b0 = s1[0], b1 = s1[1];
    f[0] = a0.x; f[1] = a0.y; f[2] = a0.z; f[3] = a0.w;
    f[4] = a1.x; f[5] = a1.y; f[6] = a1.z; f[7] = a1.w;
    f[8]  = b0.x; f[9]  = b0.y; f[10] = b0.z; f[11] = b0.w;
    f[12] = b1.x; f[13] = b1.y; f[14] = b1.z; f[15] = b1.w;
}

// ---------------------------------------------------------------------------
// Kernel 1: fused energy kernel. One block per (b,s). 256 threads (8 waves).
// out[a,p] = sum_{c'=0..511} W1[a,c'] * X[c',p],  X = [hidden[b]; enc[b,s]]
// then energy[b,s] = mean_p( sum_a W2[a]*tanh(out[a,p]+b1[a]) ) + b2.
// Split-bf16 GEMM: acc += Ah*Bh + Ah*Bl + Al*Bh  (fp32-quality).
// Software-pipelined: global prefetch of next chunk overlaps WMMA phase.
// ---------------------------------------------------------------------------
__global__ __launch_bounds__(256, 1) void energy_kernel(
    const float* __restrict__ hidden, const float* __restrict__ enc,
    const unsigned short* __restrict__ fragHi,
    const unsigned short* __restrict__ fragLo,
    const float* __restrict__ b1, const float* __restrict__ W2,
    const float* __restrict__ b2, float* __restrict__ energy_out)
{
    __shared__ __align__(16) unsigned short sXhi[128 * LDK];  // transposed [p][k] bf16
    __shared__ __align__(16) unsigned short sXlo[128 * LDK];
    __shared__ float sEnergy[256];
    __shared__ float sB1[256];
    __shared__ float sW2[256];
    __shared__ float sRed[8];

    int blk = blockIdx.x;
    int b = blk >> 5, s = blk & 31;
    int tid = threadIdx.x;
    int lane = tid & 31, wave = tid >> 5;

    sEnergy[tid] = 0.0f;
    sB1[tid] = b1[tid];
    sW2[tid] = W2[tid];
    __syncthreads();

    // per-lane epilogue constants: a = (wave*2+i)*16 + v + 8*(lane>>4)
    float b1v[16], w2v[16];
#pragma unroll
    for (int i = 0; i < 2; ++i)
#pragma unroll
        for (int v = 0; v < 8; ++v) {
            int a = (wave * 2 + i) * 16 + v + 8 * (lane >> 4);
            b1v[i * 8 + v] = sB1[a];
            w2v[i * 8 + v] = sW2[a];
        }

    int kpair = tid >> 4;      // 0..15 -> k rows 2*kpair, 2*kpair+1
    int cseg  = tid & 15;      // 8-pixel column segment
    int prow  = lane & 15;
    int kb    = (lane >> 4) * 16;

    for (int ph = 0; ph < 2; ++ph) {           // pixel halves [0,128) / [128,256)
        v8f acc[2][8];
#pragma unroll
        for (int i = 0; i < 2; ++i)
#pragma unroll
            for (int p = 0; p < 8; ++p) acc[i][p] = (v8f){0,0,0,0,0,0,0,0};

        float fA[16], fB[16];
        load_chunk(hidden, enc, b, s, ph, 0, kpair, cseg, fA);

#pragma unroll 1
        for (int ks2 = 0; ks2 < 16; ks2 += 2) {
#pragma unroll
            for (int half = 0; half < 2; ++half) {
                int ks = ks2 + half;
                float* fcur = half ? fB : fA;
                float* fnxt = half ? fA : fB;

                // ---- store current chunk -> LDS (bf16 hi/lo, transposed,
                //      k-pairs packed into dwords; all indices compile-time) ----
#pragma unroll
                for (int j = 0; j < 8; ++j) {
                    float x0 = fcur[j];        // k = 2*kpair
                    float x1 = fcur[j + 8];    // k = 2*kpair+1
                    float h0 = bf16_to_f32(f32_to_bf16(x0));
                    float h1 = bf16_to_f32(f32_to_bf16(x1));
                    unsigned hi = pack_bf16x2(x0, x1);
                    unsigned lo = pack_bf16x2(x0 - h0, x1 - h1);
                    int idx = (cseg * 8 + j) * LDK + 2 * kpair;   // even -> dword-aligned
                    *(unsigned*)&sXhi[idx] = hi;
                    *(unsigned*)&sXlo[idx] = lo;
                }
                __syncthreads();

                // ---- prefetch next chunk (overlaps WMMA phase) ----
                load_chunk(hidden, enc, b, s, ph, (ks + 1) & 15, kpair, cseg, fnxt);

                // ---- A fragments (global, L2-resident) ----
                v16bf aHi[2], aLo[2];
#pragma unroll
                for (int i = 0; i < 2; ++i) {
                    int fb = (((wave * 2 + i) * 16 + ks) * 32 + lane) * 16;
                    aHi[i] = *(const v16bf*)(fragHi + fb);
                    aLo[i] = *(const v16bf*)(fragLo + fb);
                }

                // ---- WMMA: B frag = 16 consecutive bf16 per lane from LDS ----
#pragma unroll
                for (int pt = 0; pt < 8; ++pt) {
                    int off = (pt * 16 + prow) * LDK + kb;
                    v16bf bHi = *(const v16bf*)(sXhi + off);
                    v16bf bLo = *(const v16bf*)(sXlo + off);
#pragma unroll
                    for (int i = 0; i < 2; ++i) {
                        acc[i][pt] = __builtin_amdgcn_wmma_f32_16x16x32_bf16(
                            false, aHi[i], false, bHi, (short)0, acc[i][pt], false, false);
                        acc[i][pt] = __builtin_amdgcn_wmma_f32_16x16x32_bf16(
                            false, aHi[i], false, bLo, (short)0, acc[i][pt], false, false);
                        acc[i][pt] = __builtin_amdgcn_wmma_f32_16x16x32_bf16(
                            false, aLo[i], false, bHi, (short)0, acc[i][pt], false, false);
                    }
                }
                __syncthreads();
            }
        }

        // -------- epilogue: tanh + W2-dot, per-pixel reduce into LDS --------
#pragma unroll
        for (int pt = 0; pt < 8; ++pt) {
            float psum = 0.0f;
#pragma unroll
            for (int i = 0; i < 2; ++i)
#pragma unroll
                for (int v = 0; v < 8; ++v) {
                    float t = tanh_fast(acc[i][pt][v] + b1v[i * 8 + v]);
                    psum += w2v[i * 8 + v] * t;
                }
            atomicAdd(&sEnergy[ph * 128 + pt * 16 + (lane & 15)], psum);
        }
        __syncthreads();
    }

    // -------- mean over 256 pixels --------
    float vsum = sEnergy[tid];
#pragma unroll
    for (int o = 16; o >= 1; o >>= 1) vsum += __shfl_xor(vsum, o, 32);
    if (lane == 0) sRed[wave] = vsum;
    __syncthreads();
    if (tid == 0) {
        float tot = 0.0f;
#pragma unroll
        for (int w = 0; w < 8; ++w) tot += sRed[w];
        energy_out[blk] = tot * (1.0f / 256.0f) + b2[0];
    }
}

// ---------------------------------------------------------------------------
// Kernel 2: softmax over S per batch (16 blocks x 32 lanes)
// ---------------------------------------------------------------------------
__global__ __launch_bounds__(32) void softmax_kernel(
    const float* __restrict__ energy, float* __restrict__ attn)
{
    int b = blockIdx.x, t = threadIdx.x;
    float e = energy[b * NS + t];
    float m = e;
#pragma unroll
    for (int o = 16; o >= 1; o >>= 1) m = fmaxf(m, __shfl_xor(m, o, 32));
    float ex = __expf(e - m);
    float sum = ex;
#pragma unroll
    for (int o = 16; o >= 1; o >>= 1) sum += __shfl_xor(sum, o, 32);
    attn[b * NS + t] = ex / sum;
}

// ---------------------------------------------------------------------------
// Kernel 3: context = sum_s attn[b,s] * enc[b,s,c,hw]  (pure streaming;
// enc is likely still L2-resident after the energy pass)
// ---------------------------------------------------------------------------
__global__ __launch_bounds__(256) void context_kernel(
    const float* __restrict__ enc, const float* __restrict__ attn,
    float* __restrict__ out)
{
    __shared__ float sAttn[NS];
    int b = blockIdx.x >> 6;
    int idx = (blockIdx.x & 63) * 256 + threadIdx.x;  // 0..16383 per batch
    if (threadIdx.x < NS) sAttn[threadIdx.x] = attn[b * NS + threadIdx.x];
    __syncthreads();
    int c = idx >> 6, p4 = idx & 63;
    const float4* e4 = (const float4*)enc;
    float4 acc = make_float4(0.f, 0.f, 0.f, 0.f);
#pragma unroll 8
    for (int s = 0; s < NS; ++s) {
        float w = sAttn[s];
        float4 x = e4[(((size_t)(b * NS + s) * HD + c) << 6) + p4];
        acc.x += w * x.x; acc.y += w * x.y; acc.z += w * x.z; acc.w += w * x.w;
    }
    ((float4*)out)[((size_t)(b * HD + c) << 6) + p4] = acc;
}

// ---------------------------------------------------------------------------
extern "C" void kernel_launch(void* const* d_in, const int* in_sizes, int n_in,
                              void* d_out, int out_size, void* d_ws, size_t ws_size,
                              hipStream_t stream)
{
    const float* hidden = (const float*)d_in[0];
    const float* enc    = (const float*)d_in[1];
    const float* W1     = (const float*)d_in[2];
    const float* b1     = (const float*)d_in[3];
    const float* W2     = (const float*)d_in[4];
    const float* b2     = (const float*)d_in[5];
    float* out = (float*)d_out;

    // workspace layout: [fragHi 256KB][fragLo 256KB][energy 2KB][attn 2KB]
    unsigned short* fragHi = (unsigned short*)d_ws;
    unsigned short* fragLo = fragHi + 16 * 16 * 32 * 16;
    float* energy = (float*)((char*)d_ws + (size_t)2 * 16 * 16 * 32 * 16 * sizeof(unsigned short));
    float* attn = energy + NB * NS;

    prep_w1_frags<<<256, 32, 0, stream>>>(W1, fragHi, fragLo);
    energy_kernel<<<NB * NS, 256, 0, stream>>>(hidden, enc, fragHi, fragLo, b1, W2, b2, energy);
    softmax_kernel<<<NB, 32, 0, stream>>>(energy, attn);
    context_kernel<<<NB * 64, 256, 0, stream>>>(enc, attn, out);
}
